// HeteroForecastGATv2Conv_85822036509294
// MI455X (gfx1250) — matrix operand order repaired and checked by reading, not compiled, not run on hardware
//
#include <hip/hip_runtime.h>

#define HID 128
#define HEADS 4
#define TSTEPS 24
#define GATES 512
#define SLOPE 0.2f

typedef __attribute__((ext_vector_type(16))) __bf16 v16bf;
typedef __attribute__((ext_vector_type(8)))  float  v8f;

union ABFrag { v16bf v; unsigned int u[8]; };

__device__ __forceinline__ unsigned short f2bf(float f) {
  unsigned int u = __float_as_uint(f);
  u += 0x7FFFu + ((u >> 16) & 1u);           // round-to-nearest-even
  return (unsigned short)(u >> 16);
}
__device__ __forceinline__ float bf2f(unsigned short s) {
  return __uint_as_float(((unsigned int)s) << 16);
}

// ---- native-transcendental fast math (v_exp_f32 / v_rcp_f32 / v_tanh_f32) --
__device__ __forceinline__ float fexp(float x) {         // e^x, no libm fixup
  return __builtin_amdgcn_exp2f(x * 1.44269504088896340736f);
}
__device__ __forceinline__ float frcp(float x) {
  return __builtin_amdgcn_rcpf(x);
}
__device__ __forceinline__ float fsigmoid(float x) {     // 1/(1+e^-x)
  return frcp(1.0f + fexp(-x));
}
#if defined(__has_builtin)
#if __has_builtin(__builtin_amdgcn_tanhf)
#define HAVE_AMDGCN_TANH 1
#endif
#endif
__device__ __forceinline__ float ftanh(float x) {
#ifdef HAVE_AMDGCN_TANH
  return __builtin_amdgcn_tanhf(x);                      // V_TANH_F32
#else
  const float e = fexp(2.0f * x);
  return (e - 1.0f) * frcp(e + 1.0f);
#endif
}

// ---- WMMA fragment loaders (contiguous k = 16*(lane/16) + s, same order in A and B) ----------------------------
// A: 16x32 bf16 (MxK). Xb is LDS [16][HID] bf16, row-major.
__device__ __forceinline__ v16bf load_frag_a(const unsigned short* Xb, int kb, int lane) {
  const int m  = lane & 15;
  const int hi = lane >> 4;
  const unsigned short* row = Xb + m * HID + kb + hi * 16;
  ABFrag r;
  *(uint2*)&r.u[0] = *(const uint2*)(row + 0);   // K = base+0..3
  *(uint2*)&r.u[2] = *(const uint2*)(row + 4);   // K = base+4..7
  *(uint2*)&r.u[4] = *(const uint2*)(row + 8);   // K = base+8..11
  *(uint2*)&r.u[6] = *(const uint2*)(row + 12);  // K = base+12..15
  return r.v;
}
// B: 32x16 bf16 (KxN). Wt is LDS storing B transposed: Wt[n*ldk + k] = B[k][n].
__device__ __forceinline__ v16bf load_frag_b(const unsigned short* Wt, int ldk,
                                             int jt, int kb, int lane) {
  const int n  = (jt << 4) + (lane & 15);
  const int hi = lane >> 4;
  const unsigned short* p = Wt + n * ldk + kb + hi * 16;
  ABFrag r;
  *(uint4*)&r.u[0] = *(const uint4*)(p + 0);
  *(uint4*)&r.u[4] = *(const uint4*)(p + 8);
  return r.v;
}
__device__ __forceinline__ v8f wmma_bf16(v16bf a, v16bf b, v8f c) {
  return __builtin_amdgcn_wmma_f32_16x16x32_bf16(false, a, false, b, (short)0, c,
                                                 false, false);
}

// f32 -> bf16 global convert (one-shot per LSTM so the weight image can be
// DMA'd into LDS with async b128 copies, no per-element VALU convert in-block)
__global__ void __launch_bounds__(256)
convert_bf16_kernel(const float* __restrict__ src, unsigned short* __restrict__ dst, int n) {
  int i = blockIdx.x * 256 + threadIdx.x;
  if (i < n) dst[i] = f2bf(src[i]);
}

// ---- LSTM: input_size=1, hidden=128, T=24; one 16-node tile per block ------
// 256 threads = 8 waves; wave w owns hidden columns [16w,16w+16).
// Whh (B matrix, 512x128 bf16) async-DMA'd to LDS, then the 16 loop-invariant
// B fragments are hoisted into VGPRs before the time loop: the hot loop is
// 8x ds_load_b128 (recurrent h) + 16x v_wmma back-to-back + branchless
// native-trans gate math (co-executes with the XDL pipe).
__global__ void __launch_bounds__(256)
lstm_kernel(const float* __restrict__ x, const float* __restrict__ Wih,
            const unsigned short* __restrict__ WhhBf,
            const float* __restrict__ bih, const float* __restrict__ bhh,
            float* __restrict__ h_out, int N) {
  __shared__ __align__(16) unsigned short Wt[GATES * HID];   // 128 KB bf16
  __shared__ __align__(16) unsigned short hbuf[16 * HID];    // 4 KB bf16
  __shared__ float xt_s[16 * TSTEPS];
  __shared__ float wih_s[GATES];
  __shared__ float bias_s[GATES];

  const int tid  = threadIdx.x;
  const int lane = tid & 31;
  const int wave = tid >> 5;
  const int nb   = blockIdx.x;

  // Async DMA of the 128KB bf16 weight image into LDS (GLOBAL_LOAD_ASYNC_TO_LDS,
  // tracked by ASYNCcnt).  Per-lane b128 => 512B per wave-instruction.
  {
    const unsigned ldsbase = (unsigned)(size_t)(&Wt[0]);   // addr[31:0] = LDS offset
#pragma unroll
    for (int it = 0; it < (GATES * HID * 2) / (256 * 16); ++it) {  // 32 iters
      unsigned off = (unsigned)(it * 4096 + tid * 16);
      unsigned lds_off = ldsbase + off;
      asm volatile("global_load_async_to_lds_b128 %0, %1, %2"
                   :: "v"(lds_off), "v"(off), "s"(WhhBf)
                   : "memory");
    }
  }
  for (int i = tid; i < GATES; i += 256) { wih_s[i] = Wih[i]; bias_s[i] = bih[i] + bhh[i]; }
  for (int i = tid; i < 16 * TSTEPS; i += 256) {
    int m = i / TSTEPS, t = i % TSTEPS, n = nb * 16 + m;
    xt_s[i] = (n < N) ? x[n * TSTEPS + t] : 0.0f;
  }
  for (int i = tid; i < 16 * HID; i += 256) hbuf[i] = 0;
  asm volatile("s_wait_asynccnt 0x0" ::: "memory");
  __syncthreads();

  // hoist loop-invariant B fragments (Whh) into registers: [gate][kstep]
  v16bf bw[4][4];
#pragma unroll
  for (int g = 0; g < 4; ++g)
#pragma unroll
    for (int kbi = 0; kbi < 4; ++kbi)
      bw[g][kbi] = load_frag_b(Wt, HID, wave + 8 * g, kbi * 32, lane);

  const int hi   = lane >> 4;
  const int colw = lane & 15;
  const int col  = wave * 16 + colw;               // hidden index 0..127
  const float wi = wih_s[col],           bi = bias_s[col];
  const float wf = wih_s[HID + col],     bf = bias_s[HID + col];
  const float wg = wih_s[2 * HID + col], bg = bias_s[2 * HID + col];
  const float wo = wih_s[3 * HID + col], bo = bias_s[3 * HID + col];

  float cst[8];
#pragma unroll
  for (int r = 0; r < 8; ++r) cst[r] = 0.0f;

  for (int t = 0; t < TSTEPS; ++t) {
    v16bf afr[4];
#pragma unroll
    for (int kbi = 0; kbi < 4; ++kbi)
      afr[kbi] = load_frag_a(hbuf, kbi * 32, lane);

    v8f acc[4];
    const v8f vz = {0.f, 0.f, 0.f, 0.f, 0.f, 0.f, 0.f, 0.f};
#pragma unroll
    for (int g = 0; g < 4; ++g) acc[g] = vz;
#pragma unroll
    for (int kbi = 0; kbi < 4; ++kbi)
#pragma unroll
      for (int g = 0; g < 4; ++g)
        acc[g] = wmma_bf16(afr[kbi], bw[g][kbi], acc[g]);

    __syncthreads();                               // all reads of hbuf done
#pragma unroll
    for (int r = 0; r < 8; ++r) {
      const int m  = hi * 8 + r;                   // C/D layout: M = r (+8 hi lanes)
      const float xv = xt_s[m * TSTEPS + t];
      const float gi = acc[0][r] + xv * wi + bi;
      const float gf = acc[1][r] + xv * wf + bf;
      const float gg = acc[2][r] + xv * wg + bg;
      const float go = acc[3][r] + xv * wo + bo;
      const float c  = fsigmoid(gf) * cst[r] + fsigmoid(gi) * ftanh(gg);
      cst[r] = c;
      hbuf[m * HID + col] = f2bf(fsigmoid(go) * ftanh(c));
    }
    __syncthreads();
  }

  for (int i = tid; i < 16 * HID; i += 256) {
    int m = i >> 7, k = i & 127, n = nb * 16 + m;
    if (n < N) h_out[n * HID + k] = bf2f(hbuf[i]);
  }
}

// ---- Y[N,128] = X[N,128] @ W[128,128] + b  (W row-major k x j) --------------
__global__ void __launch_bounds__(128)
linear_kernel(const float* __restrict__ X, const float* __restrict__ W,
              const float* __restrict__ bias, float* __restrict__ Y, int N) {
  __shared__ __align__(16) unsigned short Wt[HID * HID];  // Wt[j][k] = W[k][j]
  __shared__ __align__(16) unsigned short Xb[16 * HID];
  const int tid  = threadIdx.x;
  const int lane = tid & 31;
  const int wave = tid >> 5;
  const int nb   = blockIdx.x;

  for (int i = tid; i < HID * HID; i += 128) {
    int j = i >> 7, k = i & 127;
    Wt[i] = f2bf(W[k * HID + j]);
  }
  for (int i = tid; i < 16 * HID; i += 128) {
    int m = i >> 7, k = i & 127, n = nb * 16 + m;
    Xb[i] = f2bf((n < N) ? X[n * HID + k] : 0.0f);
  }
  __syncthreads();

  const int hi   = lane >> 4;
  const int colw = lane & 15;
#pragma unroll
  for (int q = 0; q < 2; ++q) {
    const int jt = wave + 4 * q;
    v8f acc = {0.f, 0.f, 0.f, 0.f, 0.f, 0.f, 0.f, 0.f};
#pragma unroll
    for (int kb = 0; kb < HID; kb += 32) {
      v16bf a = load_frag_a(Xb, kb, lane);
      v16bf b = load_frag_b(Wt, HID, jt, kb, lane);
      acc = wmma_bf16(a, b, acc);
    }
    const int col = jt * 16 + colw;
    const float bv = bias[col];
#pragma unroll
    for (int r = 0; r < 8; ++r) {
      const int n = nb * 16 + hi * 8 + r;
      if (n < N) Y[n * HID + col] = acc[r] + bv;
    }
  }
}

// ---- edge-phase helpers -----------------------------------------------------
__device__ __forceinline__ void atomicMaxF(float* addr, float val) {
  if (val >= 0.0f) atomicMax((int*)addr, __float_as_int(val));
  else             atomicMin((unsigned int*)addr, (unsigned int)__float_as_int(val));
}
__device__ __forceinline__ void edge_sd(const int* __restrict__ src,
                                        const int* __restrict__ dst,
                                        int e, int E, int swap, int& s, int& d) {
  if (e < E) {
    s = src[e]; d = dst[e];
    if (swap) { int t = s; s = d; d = t; }
  } else {                 // appended self loop
    s = d = e - E;
  }
}

__global__ void init_attn_kernel(float* __restrict__ mmax, float* __restrict__ denom, int n) {
  int i = blockIdx.x * 256 + threadIdx.x;
  if (i < n) { mmax[i] = -__builtin_huge_valf(); denom[i] = 0.0f; }
}
__global__ void zero_kernel(float* __restrict__ p, int n) {
  int i = blockIdx.x * 256 + threadIdx.x;
  if (i < n) p[i] = 0.0f;
}

// one wave per edge: leaky(hl[s]+hr[d]) . att  -> logit per head, segment max
__global__ void __launch_bounds__(256)
attn_logits_kernel(const float* __restrict__ hl, const float* __restrict__ hr,
                   const int* __restrict__ src, const int* __restrict__ dst,
                   int E, int nloops, int swap, const float* __restrict__ att,
                   float* __restrict__ logits, float* __restrict__ mmax) {
  const int e = blockIdx.x * 8 + (threadIdx.x >> 5);
  const int lane = threadIdx.x & 31;
  if (e >= E + nloops) return;
  int s, d; edge_sd(src, dst, e, E, swap, s, d);
  const float4 l4 = *(const float4*)&hl[(size_t)s * HID + lane * 4];
  const float4 r4 = *(const float4*)&hr[(size_t)d * HID + lane * 4];
  const float4 a4 = *(const float4*)&att[lane * 4];
  float v, acc = 0.0f;
  v = l4.x + r4.x; acc += (v > 0.f ? v : SLOPE * v) * a4.x;
  v = l4.y + r4.y; acc += (v > 0.f ? v : SLOPE * v) * a4.y;
  v = l4.z + r4.z; acc += (v > 0.f ? v : SLOPE * v) * a4.z;
  v = l4.w + r4.w; acc += (v > 0.f ? v : SLOPE * v) * a4.w;
  acc += __shfl_xor(acc, 1);
  acc += __shfl_xor(acc, 2);
  acc += __shfl_xor(acc, 4);                       // 8-lane head group reduce
  const int h = lane >> 3;
  if ((lane & 7) == 0) {
    logits[(size_t)e * HEADS + h] = acc;
    atomicMaxF(&mmax[(size_t)d * HEADS + h], acc);
  }
}

// thread per edge: a = exp(logit - m[d]); denom[d] += a; store a in-place
__global__ void __launch_bounds__(256)
attn_expsum_kernel(const int* __restrict__ src, const int* __restrict__ dst,
                   int E, int nloops, int swap, float* __restrict__ logits,
                   const float* __restrict__ mmax, float* __restrict__ denom) {
  const int e = blockIdx.x * 256 + threadIdx.x;
  if (e >= E + nloops) return;
  int s, d; edge_sd(src, dst, e, E, swap, s, d);
#pragma unroll
  for (int h = 0; h < HEADS; ++h) {
    float a = fexp(logits[(size_t)e * HEADS + h] - mmax[(size_t)d * HEADS + h]);
    logits[(size_t)e * HEADS + h] = a;
    atomicAdd(&denom[(size_t)d * HEADS + h], a);
  }
}

// one wave per edge: acc[d] += wgt * alpha * hl[s]
__global__ void __launch_bounds__(256)
attn_scatter_kernel(const float* __restrict__ hl, const int* __restrict__ src,
                    const int* __restrict__ dst, int E, int nloops, int swap,
                    const float* __restrict__ logits, const float* __restrict__ denom,
                    float* __restrict__ accb, float wgt) {
  const int e = blockIdx.x * 8 + (threadIdx.x >> 5);
  const int lane = threadIdx.x & 31;
  if (e >= E + nloops) return;
  int s, d; edge_sd(src, dst, e, E, swap, s, d);
  const int h = lane >> 3;
  const float alpha = wgt * logits[(size_t)e * HEADS + h] *
                      frcp(denom[(size_t)d * HEADS + h] + 1e-16f);
  const float4 l4 = *(const float4*)&hl[(size_t)s * HID + lane * 4];
  float* o = &accb[(size_t)d * HID + lane * 4];
  atomicAdd(o + 0, alpha * l4.x);
  atomicAdd(o + 1, alpha * l4.y);
  atomicAdd(o + 2, alpha * l4.z);
  atomicAdd(o + 3, alpha * l4.w);
}

// xfin = relu(acc + 0.25*(b1+b2) + 0.5*b3 + h_t)
__global__ void __launch_bounds__(256)
combine_kernel(const float* __restrict__ accb, const float* __restrict__ h_t,
               const float* __restrict__ b1, const float* __restrict__ b2,
               const float* __restrict__ b3, float* __restrict__ xfin, int n) {
  int i = blockIdx.x * 256 + threadIdx.x;
  if (i < n) {
    int ch = i & (HID - 1);
    float wb = 0.25f * (b1[ch] + b2[ch]) + 0.5f * b3[ch];
    float v = accb[i] + wb + h_t[i];
    xfin[i] = v > 0.f ? v : 0.f;
  }
}

// ---- host orchestration -----------------------------------------------------
extern "C" void kernel_launch(void* const* d_in, const int* in_sizes, int n_in,
                              void* d_out, int out_size, void* d_ws, size_t ws_size,
                              hipStream_t stream) {
  (void)n_in; (void)out_size; (void)ws_size;
  const float* x_t   = (const float*)d_in[0];
  const float* x_c   = (const float*)d_in[1];
  const int*   ei_tt = (const int*)d_in[2];
  const int*   ei_ct = (const int*)d_in[3];
  // params flattened in setup_inputs() insertion order:
  const float* Wih_t = (const float*)d_in[4];
  const float* Whh_t = (const float*)d_in[5];
  const float* bih_t = (const float*)d_in[6];
  const float* bhh_t = (const float*)d_in[7];
  const float* Wih_c = (const float*)d_in[8];
  const float* Whh_c = (const float*)d_in[9];
  const float* bih_c = (const float*)d_in[10];
  const float* bhh_c = (const float*)d_in[11];
  const float* Wl_f = (const float*)d_in[12]; const float* bl_f = (const float*)d_in[13];
  const float* Wr_f = (const float*)d_in[14]; const float* br_f = (const float*)d_in[15];
  const float* at_f = (const float*)d_in[16]; const float* bs_f = (const float*)d_in[17];
  const float* Wl_r = (const float*)d_in[18]; const float* bl_r = (const float*)d_in[19];
  const float* Wr_r = (const float*)d_in[20]; const float* br_r = (const float*)d_in[21];
  const float* at_r = (const float*)d_in[22]; const float* bs_r = (const float*)d_in[23];
  const float* Wl_c = (const float*)d_in[24]; const float* bl_c = (const float*)d_in[25];
  const float* Wr_c = (const float*)d_in[26]; const float* br_c = (const float*)d_in[27];
  const float* at_c = (const float*)d_in[28]; const float* bs_c = (const float*)d_in[29];
  const float* linW = (const float*)d_in[30]; const float* linb = (const float*)d_in[31];

  const int Nt  = in_sizes[0] / TSTEPS;
  const int Nc  = in_sizes[1] / TSTEPS;
  const int Ett = in_sizes[2] / 2;
  const int Ect = in_sizes[3] / 2;
  const int Nmax = Nt > Nc ? Nt : Nc;

  float* p = (float*)d_ws;
  float* h_t    = p; p += (size_t)Nt * HID;
  float* h_c    = p; p += (size_t)Nc * HID;
  float* hl     = p; p += (size_t)Nmax * HID;
  float* hr     = p; p += (size_t)Nt * HID;     // also reused as xfin after convs
  float* accb   = p; p += (size_t)Nt * HID;
  float* mmax   = p; p += (size_t)Nt * HEADS;
  float* denom  = p; p += (size_t)Nt * HEADS;
  float* logits = p; p += (size_t)(Ett + Nt) * HEADS;
  unsigned short* whhbf = (unsigned short*)p; p += (GATES * HID) / 2;  // 128KB bf16
  float* xfin   = hr;

  // 1) LSTM pretransforms (bf16 weight image -> async LDS DMA -> WMMA loop)
  convert_bf16_kernel<<<(GATES * HID + 255) / 256, 256, 0, stream>>>(Whh_t, whhbf, GATES * HID);
  lstm_kernel<<<(Nt + 15) / 16, 256, 0, stream>>>(x_t, Wih_t, whhbf, bih_t, bhh_t, h_t, Nt);
  convert_bf16_kernel<<<(GATES * HID + 255) / 256, 256, 0, stream>>>(Whh_c, whhbf, GATES * HID);
  lstm_kernel<<<(Nc + 15) / 16, 256, 0, stream>>>(x_c, Wih_c, whhbf, bih_c, bhh_c, h_c, Nc);

  // 2) shared accumulator for all three convs (mix weights folded into scatter)
  zero_kernel<<<((Nt * HID) + 255) / 256, 256, 0, stream>>>(accb, Nt * HID);

  auto run_conv = [&](const float* xsrc, int Nsrc, const float* Wl, const float* bl,
                      const float* Wr, const float* br, const float* att,
                      const int* sI, const int* dI, int E, int nloops, int swap, float w) {
    linear_kernel<<<(Nsrc + 15) / 16, 128, 0, stream>>>(xsrc, Wl, bl, hl, Nsrc);
    linear_kernel<<<(Nt + 15) / 16, 128, 0, stream>>>(h_t, Wr, br, hr, Nt);
    init_attn_kernel<<<((Nt * HEADS) + 255) / 256, 256, 0, stream>>>(mmax, denom, Nt * HEADS);
    const int Etot = E + nloops;
    attn_logits_kernel<<<(Etot + 7) / 8, 256, 0, stream>>>(hl, hr, sI, dI, E, nloops, swap,
                                                           att, logits, mmax);
    attn_expsum_kernel<<<(Etot + 255) / 256, 256, 0, stream>>>(sI, dI, E, nloops, swap,
                                                               logits, mmax, denom);
    attn_scatter_kernel<<<(Etot + 7) / 8, 256, 0, stream>>>(hl, sI, dI, E, nloops, swap,
                                                            logits, denom, accb, w);
  };

  // DirGATv2 fwd/rev on target-target edges (+self loops), weight (1-a)/2, a/2
  run_conv(h_t, Nt, Wl_f, bl_f, Wr_f, br_f, at_f, ei_tt, ei_tt + Ett, Ett, Nt, 0, 0.25f);
  run_conv(h_t, Nt, Wl_r, bl_r, Wr_r, br_r, at_r, ei_tt, ei_tt + Ett, Ett, Nt, 1, 0.25f);
  // bipartite context->target, weight 1/2
  run_conv(h_c, Nc, Wl_c, bl_c, Wr_c, br_c, at_c, ei_ct, ei_ct + Ect, Ect, 0, 0, 0.5f);

  // 3) skip + relu, then final Linear via WMMA into d_out
  combine_kernel<<<((Nt * HID) + 255) / 256, 256, 0, stream>>>(accb, h_t, bs_f, bs_r, bs_c,
                                                               xfin, Nt * HID);
  linear_kernel<<<(Nt + 15) / 16, 128, 0, stream>>>(xfin, linW, linb, (float*)d_out, Nt);
}
